// DeepSeekV3_3796751090030
// MI455X (gfx1250) — matrix operand order repaired and checked by reading, not compiled, not run on hardware
//
#include <hip/hip_runtime.h>
#include <hip/hip_bf16.h>
#include <math.h>

// ---------------- model constants ----------------
static constexpr int Vv  = 32000;
static constexpr int Hh  = 1024;
static constexpr int NL  = 2;
static constexpr int NH  = 16;
static constexpr int HD  = 64;
static constexpr int RD  = 64;
static constexpr int KVC = 256;
static constexpr int QC  = 384;
static constexpr int Ee  = 8;
static constexpr int Ii  = 4096;
static constexpr int Bb  = 2;
static constexpr int Ss  = 1024;
static constexpr int Tt  = Bb * Ss;

// ---------------- CDNA5 async-to-LDS availability ----------------
#if defined(__has_builtin)
#if __has_builtin(__builtin_amdgcn_global_load_async_to_lds_b128) && \
    __has_builtin(__builtin_amdgcn_s_wait_asynccnt)
#define USE_ASYNC_LDS 1
#endif
#endif

#ifdef USE_ASYNC_LDS
typedef int v4i_async __attribute__((vector_size(16)));
__device__ __forceinline__ void async_copy_b128(const void* gsrc, void* ldst) {
    // builtin expects generic pointers to 16B int vectors (probe: param0 is
    // 'int __attribute__((vector_size(16))) *'); backend inserts addrspace casts.
    __builtin_amdgcn_global_load_async_to_lds_b128(
        (v4i_async*)gsrc, (v4i_async*)ldst, 0, 0);
}
#endif

// ---------------- WMMA types ----------------
typedef __attribute__((ext_vector_type(16))) __bf16 v16bf;
typedef __attribute__((ext_vector_type(8)))  float  v8f;

union Frag {
    v16bf v;
    uint4 u[2];
    unsigned short h[16];
};

__device__ __forceinline__ v8f vzero8() {
    v8f z = {0.f,0.f,0.f,0.f,0.f,0.f,0.f,0.f};
    return z;
}

// hardware bf16 convert (v_cvt_*_bf16_f32 on gfx1250)
__device__ __forceinline__ unsigned short f2bf(float f) {
    union { __bf16 b; unsigned short s; } u;
    u.b = (__bf16)f;
    return u.s;
}

// A fragment (16x32 bf16, MxK): lane holds row M=lane&15,
// K chunks {k0..k0+7} and {k0+16..k0+23}, k0 = (lane<16?0:8).
__device__ __forceinline__ void load_a_frag(Frag& f, const unsigned short* base,
                                            int stride, int koff, int lane) {
    const unsigned short* p = base + (lane & 15) * stride + koff + ((lane < 16) ? 0 : 8);
    f.u[0] = *reinterpret_cast<const uint4*>(p);
    f.u[1] = *reinterpret_cast<const uint4*>(p + 16);
}

// B fragment (32x16 bf16, KxN) staged as [n][k] row-major: lane holds col N=lane&15,
// K chunks {k0..k0+7} and {k0+16..k0+23}, k0 = (lane<16?0:8).
__device__ __forceinline__ void load_b_frag(Frag& f, const unsigned short* base,
                                            int stride, int koff, int lane) {
    const unsigned short* p = base + (lane & 15) * stride + koff + ((lane < 16) ? 0 : 8);
    f.u[0] = *reinterpret_cast<const uint4*>(p);
    f.u[1] = *reinterpret_cast<const uint4*>(p + 16);
}

__device__ __forceinline__ float gelu_exact(float x) {
    return 0.5f * x * (1.0f + erff(x * 0.70710678118654752f));
}

// ======================================================================
// Generic GEMM:  C[M,N] = epi(A[M,K] @ W[N,K]^T + bias)
// Block 128 threads = 4 waves (2x2); block tile 128x128; wave tile 64x64.
// Double-buffered LDS staging (fp32 global -> bf16 LDS).
// M%128==0, N%128==0, K%32==0.
// ======================================================================
__launch_bounds__(128)
__global__ void gemm_wmma(float* __restrict__ C, const float* __restrict__ A,
                          const float* __restrict__ W, const float* __restrict__ bias,
                          int M, int N, int K, int gelu, int accum, float alpha,
                          const float* __restrict__ rowscale, int rs_stride)
{
    __shared__ __align__(16) unsigned short As[2][128][40];
    __shared__ __align__(16) unsigned short Bs[2][128][40];

    const int tid  = threadIdx.x;
    const int lane = tid & 31;
    const int wave = tid >> 5;
    const int wm   = wave >> 1;   // 0..1
    const int wn   = wave & 1;    // 0..1
    const int nblock = blockIdx.x * 128;
    const int mblock = blockIdx.y * 128;

    v8f acc[4][4];
    #pragma unroll
    for (int i = 0; i < 4; i++)
        #pragma unroll
        for (int j = 0; j < 2; j++) { acc[i][j] = vzero8(); acc[i][j+2] = vzero8(); }

    // each thread stages one full 32-elem row of A and of W per K-step
    auto stage = [&](int buf, int k0) {
        const float* ga = A + (size_t)(mblock + tid) * K + k0;
        const float* gb = W + (size_t)(nblock + tid) * K + k0;
        #pragma unroll
        for (int v = 0; v < 8; v++) {
            float4 ta = *reinterpret_cast<const float4*>(ga + v * 4);
            union { __bf16 h[4]; uint2 u; } pa;
            pa.h[0] = (__bf16)ta.x; pa.h[1] = (__bf16)ta.y;
            pa.h[2] = (__bf16)ta.z; pa.h[3] = (__bf16)ta.w;
            *reinterpret_cast<uint2*>(&As[buf][tid][v * 4]) = pa.u;
            float4 tb = *reinterpret_cast<const float4*>(gb + v * 4);
            union { __bf16 h[4]; uint2 u; } pb;
            pb.h[0] = (__bf16)tb.x; pb.h[1] = (__bf16)tb.y;
            pb.h[2] = (__bf16)tb.z; pb.h[3] = (__bf16)tb.w;
            *reinterpret_cast<uint2*>(&Bs[buf][tid][v * 4]) = pb.u;
        }
    };

    const int nk = K / 32;
    stage(0, 0);
    __syncthreads();

    for (int kt = 0; kt < nk; kt++) {
        const int buf = kt & 1;
        if (kt + 1 < nk) stage(buf ^ 1, (kt + 1) * 32);   // overlap with WMMA on buf
        if (kt + 2 < nk) {                                 // L2 prefetch hint
            __builtin_prefetch(A + (size_t)(mblock + tid) * K + (kt + 2) * 32, 0, 1);
            __builtin_prefetch(W + (size_t)(nblock + tid) * K + (kt + 2) * 32, 0, 1);
        }

        Frag af[4], bfr[4];
        #pragma unroll
        for (int mi = 0; mi < 4; mi++)
            load_a_frag(af[mi], &As[buf][wm * 64 + mi * 16][0], 40, 0, lane);
        #pragma unroll
        for (int ni = 0; ni < 4; ni++)
            load_b_frag(bfr[ni], &Bs[buf][wn * 64 + ni * 16][0], 40, 0, lane);

        #pragma unroll
        for (int mi = 0; mi < 4; mi++)
            #pragma unroll
            for (int ni = 0; ni < 4; ni++)
                acc[mi][ni] = __builtin_amdgcn_wmma_f32_16x16x32_bf16(
                    false, af[mi].v, false, bfr[ni].v,
                    (short)0, acc[mi][ni], false, false);
        __syncthreads();
    }

    // ---- epilogue: bias, gelu, scale, (accumulate), store ----
    #pragma unroll
    for (int mi = 0; mi < 4; mi++) {
        const int rowbase = mblock + wm * 64 + mi * 16 + ((lane >= 16) ? 8 : 0);
        #pragma unroll
        for (int ni = 0; ni < 4; ni++) {
            const int col = nblock + wn * 64 + ni * 16 + (lane & 15);
            if (col >= N) continue;
            const float bv = bias ? bias[col] : 0.0f;
            #pragma unroll
            for (int i = 0; i < 8; i++) {
                const int row = rowbase + i;
                float v = acc[mi][ni][i] + bv;
                if (gelu) v = gelu_exact(v);
                float s = alpha;
                if (rowscale) s *= rowscale[(size_t)row * rs_stride];
                float o = s * v;
                const size_t idx = (size_t)row * N + col;
                if (accum) o += C[idx];
                C[idx] = o;
            }
        }
    }
}

// ======================================================================
// Embedding gather
// ======================================================================
__global__ void embed_kernel(float* __restrict__ x, const float* __restrict__ emb,
                             const int* __restrict__ ids)
{
    const int t  = blockIdx.x;
    const int id = ids[t];
    const float* src = emb + (size_t)id * Hh;
    float* dst = x + (size_t)t * Hh;
    for (int i = threadIdx.x; i < Hh; i += blockDim.x) dst[i] = src[i];
}

// ======================================================================
// fp32 -> bf16 buffer conversion (8 elems / thread)
// ======================================================================
__global__ void cvt_bf16_kernel(unsigned short* __restrict__ dst,
                                const float* __restrict__ src, int n)
{
    const int i = (blockIdx.x * blockDim.x + threadIdx.x) * 8;
    if (i >= n) return;
    float4 a = *reinterpret_cast<const float4*>(src + i);
    float4 b = *reinterpret_cast<const float4*>(src + i + 4);
    union { __bf16 h[8]; uint4 u; } o;
    o.h[0] = (__bf16)a.x; o.h[1] = (__bf16)a.y; o.h[2] = (__bf16)a.z; o.h[3] = (__bf16)a.w;
    o.h[4] = (__bf16)b.x; o.h[5] = (__bf16)b.y; o.h[6] = (__bf16)b.z; o.h[7] = (__bf16)b.w;
    *reinterpret_cast<uint4*>(dst + i) = o.u;
}

// ======================================================================
// LayerNorm; mode==1 additionally does y = y - mean(y) + 1 (attn pre-norm)
// ======================================================================
__launch_bounds__(256)
__global__ void ln_kernel(float* __restrict__ out, const float* __restrict__ in,
                          const float* __restrict__ g, const float* __restrict__ b,
                          int mode)
{
    __shared__ float red[256];
    const int t = blockIdx.x, tid = threadIdx.x;
    const float* xr = in + (size_t)t * Hh;

    float lx[4];
    float s = 0.f;
    #pragma unroll
    for (int i = 0; i < 4; i++) { lx[i] = xr[tid + 256 * i]; s += lx[i]; }
    red[tid] = s; __syncthreads();
    for (int o = 128; o > 0; o >>= 1) { if (tid < o) red[tid] += red[tid + o]; __syncthreads(); }
    const float mu = red[0] * (1.f / Hh);
    __syncthreads();

    float var = 0.f;
    #pragma unroll
    for (int i = 0; i < 4; i++) { float d = lx[i] - mu; var += d * d; }
    red[tid] = var; __syncthreads();
    for (int o = 128; o > 0; o >>= 1) { if (tid < o) red[tid] += red[tid + o]; __syncthreads(); }
    const float rstd = rsqrtf(red[0] * (1.f / Hh) + 1e-5f);
    __syncthreads();

    float y[4]; float sy = 0.f;
    #pragma unroll
    for (int i = 0; i < 4; i++) {
        const int c = tid + 256 * i;
        y[i] = (lx[i] - mu) * rstd * g[c] + b[c];
        sy += y[i];
    }
    if (mode == 1) {
        red[tid] = sy; __syncthreads();
        for (int o = 128; o > 0; o >>= 1) { if (tid < o) red[tid] += red[tid + o]; __syncthreads(); }
        const float muy = red[0] * (1.f / Hh);
        #pragma unroll
        for (int i = 0; i < 4; i++) y[i] += 1.0f - muy;
    }
    #pragma unroll
    for (int i = 0; i < 4; i++) out[(size_t)t * Hh + tid + 256 * i] = y[i];
}

// ======================================================================
// RoPE in-place on (T, NH*RD) fp32 buffer; one wave per (token, head)
// ======================================================================
__launch_bounds__(32)
__global__ void rope_kernel(float* __restrict__ buf)
{
    const int bh = blockIdx.x;
    const int t  = bh / NH;
    const int h  = bh % NH;
    const int pos = t % Ss;
    const int j = threadIdx.x;           // 0..31 == RD/2 pairs
    float* p = buf + (size_t)t * (NH * RD) + h * RD;
    const float invf = powf(10000.0f, -(float)(2 * j) / (float)RD);
    float sn, cs;
    sincosf((float)pos * invf, &sn, &cs);
    const float x1 = p[2 * j], x2 = p[2 * j + 1];
    p[2 * j]     = x1 * cs - x2 * sn;
    p[2 * j + 1] = x1 * sn + x2 * cs;
}

// ======================================================================
// MoE routing: sigmoid(x@gate^T + bias), top-2, normalized -> dense coeff[T][E]
// ======================================================================
__launch_bounds__(32)
__global__ void route_kernel(float* __restrict__ coeff, const float* __restrict__ xn,
                             const float* __restrict__ gate, const float* __restrict__ gbias)
{
    const int t = blockIdx.x, lane = threadIdx.x;
    const float* x = xn + (size_t)t * Hh;
    float acc[Ee];
    #pragma unroll
    for (int e = 0; e < Ee; e++) acc[e] = 0.f;
    for (int i = lane; i < Hh; i += 32) {
        const float xv = x[i];
        #pragma unroll
        for (int e = 0; e < Ee; e++) acc[e] += xv * gate[(size_t)e * Hh + i];
    }
    #pragma unroll
    for (int e = 0; e < Ee; e++)
        for (int m = 16; m > 0; m >>= 1) acc[e] += __shfl_xor(acc[e], m, 32);

    if (lane == 0) {
        float sc[Ee];
        #pragma unroll
        for (int e = 0; e < Ee; e++) sc[e] = 1.0f / (1.0f + expf(-(acc[e] + gbias[e])));
        int i1 = 0, i2 = -1; float v1 = -1e30f, v2 = -1e30f;
        #pragma unroll
        for (int e = 0; e < Ee; e++) {
            if (sc[e] > v1) { v2 = v1; i2 = i1; v1 = sc[e]; i1 = e; }
            else if (sc[e] > v2) { v2 = sc[e]; i2 = e; }
        }
        const float denom = v1 + v2 + 1e-6f;
        #pragma unroll
        for (int e = 0; e < Ee; e++) coeff[(size_t)t * Ee + e] = 0.f;
        coeff[(size_t)t * Ee + i1] = v1 / denom;
        coeff[(size_t)t * Ee + i2] = v2 / denom;
    }
}

// ======================================================================
// Flash-style MLA attention over pre-converted bf16 q/k/v.
// Block = 128 threads = 4 waves; block covers 64 queries of one (b,h);
// all 4 waves share one K/V LDS staging per 32-key chunk.
// K staged via CDNA5 async global->LDS copies when available.
// ======================================================================
__launch_bounds__(128)
__global__ void attn_kernel(float* __restrict__ ctx,
                            const unsigned short* __restrict__ qch,
                            const unsigned short* __restrict__ qrh,
                            const unsigned short* __restrict__ kch,
                            const unsigned short* __restrict__ krh,
                            const unsigned short* __restrict__ vvh)
{
    __shared__ __align__(16) unsigned short Ks[32][136];   // [key][dim 0..127]
    __shared__ __align__(16) unsigned short Vt[64][40];    // [dim][key 0..31]
    __shared__ __align__(16) unsigned short Ps[4][16][40]; // per-wave P tile

    const int tid  = threadIdx.x;
    const int lane = tid & 31;
    const int w    = tid >> 5;
    int blk = blockIdx.x;
    const int qb = blk % (Ss / 64); blk /= (Ss / 64);
    const int h  = blk % NH;
    const int b  = blk / NH;
    const int qbase = qb * 64 + w * 16;

    // ---- preload Q fragments directly from bf16 global ----
    Frag qf[4];
    {
        const size_t trow = (size_t)(b * Ss + qbase + (lane & 15));
        const unsigned short* pq0 = qch + trow * (NH * HD) + h * HD;
        const unsigned short* pq1 = qrh + trow * (NH * RD) + h * RD;
        #pragma unroll
        for (int ks = 0; ks < 4; ks++) {
            const unsigned short* p = ((ks < 2) ? pq0 : pq1)
                                      + (ks & 1) * 32 + ((lane < 16) ? 0 : 8);
            qf[ks].u[0] = *reinterpret_cast<const uint4*>(p);
            qf[ks].u[1] = *reinterpret_cast<const uint4*>(p + 16);
        }
    }

    float m_i[8], l_i[8];
    v8f o[4];
    #pragma unroll
    for (int i = 0; i < 8; i++) { m_i[i] = -1e30f; l_i[i] = 0.f; }
    #pragma unroll
    for (int nt = 0; nt < 4; nt++) o[nt] = vzero8();

    const float scale = 0.08838834764831845f;   // 1/sqrt(HD+RD)
    const int nchunks = qb * 2 + 2;             // keys <= qb*64+63

    for (int c = 0; c < nchunks; c++) {
        const int base_t = b * Ss + c * 32;
        // ---- stage K rows (raw bf16 copy; async path on CDNA5) ----
        #pragma unroll
        for (int i = 0; i < 4; i++) {
            const int ch   = tid + i * 128;      // 0..511, 16B chunks
            const int part = ch >> 8;            // 0 = kc, 1 = kr
            const int key  = (ch & 255) >> 3;
            const int off  = (ch & 7) * 8;       // elems
            const unsigned short* gsrc =
                (part ? krh : kch) + (size_t)(base_t + key) * (NH * 64) + h * 64 + off;
            unsigned short* ldst = &Ks[key][part * 64 + off];
#ifdef USE_ASYNC_LDS
            async_copy_b128(gsrc, ldst);
#else
            *reinterpret_cast<uint4*>(ldst) = *reinterpret_cast<const uint4*>(gsrc);
#endif
        }
        // ---- stage V transposed: Vt[dim][key] ----
        {
            const int key = tid >> 2;
            const int seg = (tid & 3) * 16;
            const unsigned short* pv = vvh + (size_t)(base_t + key) * (NH * HD) + h * HD + seg;
            uint4 v0 = *reinterpret_cast<const uint4*>(pv);
            uint4 v1 = *reinterpret_cast<const uint4*>(pv + 8);
            const unsigned short* h0 = reinterpret_cast<const unsigned short*>(&v0);
            const unsigned short* h1 = reinterpret_cast<const unsigned short*>(&v1);
            #pragma unroll
            for (int d = 0; d < 8; d++) Vt[seg + d][key]     = h0[d];
            #pragma unroll
            for (int d = 0; d < 8; d++) Vt[seg + 8 + d][key] = h1[d];
        }
#ifdef USE_ASYNC_LDS
        __builtin_amdgcn_s_wait_asynccnt(0);
#endif
        __syncthreads();

        // ---- scores: Q(16x128) @ K^T -> two 16x16 tiles ----
        v8f s0 = vzero8(), s1 = vzero8();
        #pragma unroll
        for (int ks = 0; ks < 4; ks++) {
            Frag kf0, kf1;
            load_b_frag(kf0, &Ks[0][0],  136, ks * 32, lane);
            load_b_frag(kf1, &Ks[16][0], 136, ks * 32, lane);
            s0 = __builtin_amdgcn_wmma_f32_16x16x32_bf16(false, qf[ks].v, false, kf0.v,
                                                         (short)0, s0, false, false);
            s1 = __builtin_amdgcn_wmma_f32_16x16x32_bf16(false, qf[ks].v, false, kf1.v,
                                                         (short)0, s1, false, false);
        }

        // ---- scale + causal mask + online softmax ----
        const int colbase = c * 32 + (lane & 15);
        const int rbase   = qbase + ((lane >= 16) ? 8 : 0);
        float a0[8], a1[8], alpha_i[8];
        #pragma unroll
        for (int i = 0; i < 8; i++) {
            const int qrow = rbase + i;
            float x0 = s0[i] * scale;
            float x1 = s1[i] * scale;
            if (colbase > qrow)      x0 = -1e9f;
            if (colbase + 16 > qrow) x1 = -1e9f;
            a0[i] = x0; a1[i] = x1;
        }
        #pragma unroll
        for (int i = 0; i < 8; i++) {
            float mx = fmaxf(a0[i], a1[i]);
            for (int m = 8; m > 0; m >>= 1) mx = fmaxf(mx, __shfl_xor(mx, m, 32));
            const float mnew = fmaxf(m_i[i], mx);
            const float al   = expf(m_i[i] - mnew);
            const float p0   = expf(a0[i] - mnew);
            const float p1   = expf(a1[i] - mnew);
            float rs = p0 + p1;
            for (int m = 8; m > 0; m >>= 1) rs += __shfl_xor(rs, m, 32);
            l_i[i] = l_i[i] * al + rs;
            m_i[i] = mnew;
            alpha_i[i] = al;
            a0[i] = p0; a1[i] = p1;
        }
        #pragma unroll
        for (int nt = 0; nt < 4; nt++)
            #pragma unroll
            for (int i = 0; i < 8; i++) o[nt][i] *= alpha_i[i];

        // ---- P (C-layout) -> LDS -> A-fragment layout ----
        {
            const int prow = (lane >= 16) ? 8 : 0;
            const int pcol = lane & 15;
            #pragma unroll
            for (int i = 0; i < 8; i++) {
                Ps[w][prow + i][pcol]      = f2bf(a0[i]);
                Ps[w][prow + i][16 + pcol] = f2bf(a1[i]);
            }
        }
        __syncthreads();

        Frag pf;
        load_a_frag(pf, &Ps[w][0][0], 40, 0, lane);

        // ---- ctx += P(16x32) @ V(32x64) ----
        #pragma unroll
        for (int nt = 0; nt < 4; nt++) {
            Frag vf;
            load_b_frag(vf, &Vt[nt * 16][0], 40, 0, lane);
            o[nt] = __builtin_amdgcn_wmma_f32_16x16x32_bf16(false, pf.v, false, vf.v,
                                                            (short)0, o[nt], false, false);
        }
        __syncthreads();   // WAR before restaging Ks/Vt/Ps
    }

    // ---- write ctx (B,S,NH*HD) fp32 ----
    #pragma unroll
    for (int nt = 0; nt < 4; nt++) {
        const int d = nt * 16 + (lane & 15);
        #pragma unroll
        for (int i = 0; i < 8; i++) {
            const int row = qbase + i + ((lane >= 16) ? 8 : 0);
            ctx[(size_t)(b * Ss + row) * (NH * HD) + h * HD + d] = o[nt][i] / l_i[i];
        }
    }
}

// ======================================================================
// Host-side orchestration
// ======================================================================
extern "C" void kernel_launch(void* const* d_in, const int* in_sizes, int n_in,
                              void* d_out, int out_size, void* d_ws, size_t ws_size,
                              hipStream_t stream)
{
    (void)in_sizes; (void)n_in; (void)out_size; (void)ws_size;
    auto F = [&](int i) -> const float* { return (const float*)d_in[i]; };
    const int*   ids = (const int*)d_in[0];
    const float* emb = F(1);

    // ---- workspace bump allocator ----
    char* wp = (char*)d_ws;
    auto allocB = [&](size_t bytes) -> void* {
        void* p = (void*)wp;
        wp += ((bytes + 255) & ~(size_t)255);
        return p;
    };
    auto alloc = [&](size_t nf) -> float* { return (float*)allocB(nf * sizeof(float)); };

    float* x     = alloc((size_t)Tt * Hh);
    float* xn    = alloc((size_t)Tt * Hh);
    float* kv    = alloc((size_t)Tt * KVC);
    float* kcb   = alloc((size_t)Tt * NH * HD);
    float* vvb   = alloc((size_t)Tt * NH * HD);
    float* krb   = alloc((size_t)Tt * NH * RD);
    float* qd    = alloc((size_t)Tt * QC);
    float* qcb   = alloc((size_t)Tt * NH * HD);
    float* qrb   = alloc((size_t)Tt * NH * RD);
    float* ctx   = alloc((size_t)Tt * NH * HD);
    float* coeff = alloc((size_t)Tt * Ee);
    float* hmid  = alloc((size_t)Tt * Ii);
    float* proj  = alloc((size_t)Tt * Hh);
    unsigned short* kch = (unsigned short*)allocB((size_t)Tt * NH * HD * 2);
    unsigned short* vvh = (unsigned short*)allocB((size_t)Tt * NH * HD * 2);
    unsigned short* krh = (unsigned short*)allocB((size_t)Tt * NH * RD * 2);
    unsigned short* qch = (unsigned short*)allocB((size_t)Tt * NH * HD * 2);
    unsigned short* qrh = (unsigned short*)allocB((size_t)Tt * NH * RD * 2);

    auto gemm = [&](float* C, const float* A, const float* W, const float* bias,
                    int M, int N, int K, int gelu, int accum, float alpha,
                    const float* rs, int rss) {
        dim3 grid(N / 128, M / 128);
        gemm_wmma<<<grid, dim3(128), 0, stream>>>(C, A, W, bias, M, N, K,
                                                  gelu, accum, alpha, rs, rss);
    };
    auto cvt = [&](unsigned short* dst, const float* src, int n) {
        cvt_bf16_kernel<<<n / (256 * 8), 256, 0, stream>>>(dst, src, n);
    };

    embed_kernel<<<Tt, 256, 0, stream>>>(x, emb, ids);

    for (int l = 0; l < NL; l++) {
        const int Lb = 2 + l * 30;
        // --- MLA attention ---
        ln_kernel<<<Tt, 256, 0, stream>>>(xn, x, F(Lb + 0), F(Lb + 1), 1);
        gemm(kv,  xn, F(Lb + 2),  F(Lb + 3),  Tt, KVC,     Hh,  0, 0, 1.f, nullptr, 0);
        gemm(kcb, kv, F(Lb + 4),  F(Lb + 5),  Tt, NH * HD, KVC, 0, 0, 1.f, nullptr, 0);
        gemm(vvb, kv, F(Lb + 6),  F(Lb + 7),  Tt, NH * HD, KVC, 0, 0, 1.f, nullptr, 0);
        gemm(krb, kv, F(Lb + 8),  F(Lb + 9),  Tt, NH * RD, KVC, 0, 0, 1.f, nullptr, 0);
        gemm(qd,  xn, F(Lb + 10), F(Lb + 11), Tt, QC,      Hh,  0, 0, 1.f, nullptr, 0);
        gemm(qcb, qd, F(Lb + 12), F(Lb + 13), Tt, NH * HD, QC,  0, 0, 1.f, nullptr, 0);
        gemm(qrb, qd, F(Lb + 14), F(Lb + 15), Tt, NH * RD, QC,  0, 0, 1.f, nullptr, 0);
        rope_kernel<<<Tt * NH, 32, 0, stream>>>(krb);
        rope_kernel<<<Tt * NH, 32, 0, stream>>>(qrb);
        cvt(kch, kcb, Tt * NH * HD);
        cvt(vvh, vvb, Tt * NH * HD);
        cvt(krh, krb, Tt * NH * RD);
        cvt(qch, qcb, Tt * NH * HD);
        cvt(qrh, qrb, Tt * NH * RD);
        attn_kernel<<<Bb * NH * (Ss / 64), 128, 0, stream>>>(ctx, qch, qrh, kch, krh, vvh);
        gemm(x, ctx, F(Lb + 16), F(Lb + 17), Tt, Hh, NH * HD, 0, 1, 1.f, nullptr, 0);

        // --- MoE ---
        ln_kernel<<<Tt, 256, 0, stream>>>(xn, x, F(Lb + 18), F(Lb + 19), 0);
        route_kernel<<<Tt, 32, 0, stream>>>(coeff, xn, F(Lb + 20), F(Lb + 21));
        for (int e = 0; e < Ee; e++) {
            gemm(hmid, xn, F(Lb + 26) + (size_t)e * Ii * Hh, F(Lb + 27) + (size_t)e * Ii,
                 Tt, Ii, Hh, 1, 0, 1.f, nullptr, 0);
            gemm(x, hmid, F(Lb + 28) + (size_t)e * Hh * Ii, F(Lb + 29) + (size_t)e * Hh,
                 Tt, Hh, Ii, 0, 1, 1.f, coeff + e, Ee);
        }
        // shared expert (scaled 0.1)
        gemm(hmid, xn, F(Lb + 22), F(Lb + 23), Tt, Ii, Hh, 1, 0, 1.f,  nullptr, 0);
        gemm(x,  hmid, F(Lb + 24), F(Lb + 25), Tt, Hh, Ii, 0, 1, 0.1f, nullptr, 0);
    }

    // --- final norm, heads ---
    const int base = 2 + NL * 30;  // 62
    float* logits = (float*)d_out;
    ln_kernel<<<Tt, 256, 0, stream>>>(xn, x, F(base + 0), F(base + 1), 0);
    gemm(logits, xn, F(base + 2), F(base + 3), Tt, Vv, Hh, 0, 0, 1.f, nullptr, 0);

    gemm(proj, xn, F(base + 4), F(base + 5), Tt, Hh, Hh, 0, 0, 1.f, nullptr, 0);
    ln_kernel<<<Tt, 256, 0, stream>>>(xn, proj, F(base + 6), F(base + 7), 0);
    gemm(logits + (size_t)Tt * Vv, xn, F(base + 8), F(base + 9), Tt, Vv, Hh,
         0, 0, 1.f, nullptr, 0);
}
